// SampleWeightedFocalContrastiveLoss_16054587753049
// MI455X (gfx1250) — matrix operand order repaired and checked by reading, not compiled, not run on hardware
//
#include <hip/hip_runtime.h>

// ---------------------------------------------------------------------------
// SampleWeightedFocalContrastiveLoss for MI455X (gfx1250, wave32, WMMA)
//
// Two-pass flash-softmax over the 8192x8192 similarity matrix in 16x16 tiles
// with v_wmma_f32_16x16x32_bf16. bf16 feature copy (16 MB) stays L2-resident.
// This revision: ROLLED K-loop (unroll 1) with loop-carried bA buffers and
// iteration-local bB buffers -> compiler cannot coalesce the pipeline, keeps
// ~100 VGPRs (high occupancy) and partial s_wait_loadcnt (4 loads in flight).
// A-fragments stream from LDS each iteration (ds_load_b128, 320KB LDS path).
// ---------------------------------------------------------------------------

#define NN   8192
#define DD   1024
#define DDP  1032           // LDS row stride (+8 bf16 = 16B pad -> no bank conflicts)
#define NCLS 7
#define INV_TEMP (1.0f / 0.07f)
#define NEG_INF_V (-1.0e9f)

#define GLOBAL_AS __attribute__((address_space(1)))
#define LDS_AS    __attribute__((address_space(3)))

#if defined(__has_builtin)
#if __has_builtin(__builtin_amdgcn_global_load_async_to_lds_b128) && \
    __has_builtin(__builtin_amdgcn_s_wait_asynccnt)
#define USE_ASYNC_LDS 1
#endif
#endif

typedef __attribute__((ext_vector_type(16))) __bf16 v16bf;
typedef __attribute__((ext_vector_type(8)))  __bf16 v8bf;
typedef __attribute__((ext_vector_type(8)))  float  v8f;
typedef __attribute__((ext_vector_type(4)))  int    v4i;

union FragBF { v16bf v; v8bf h[2]; };

__device__ inline v8f wmma_bf16(const FragBF& a, const FragBF& b, v8f c) {
    return __builtin_amdgcn_wmma_f32_16x16x32_bf16(
        false, a.v, false, b.v, (short)0, c, false, false);
}

__device__ inline void ld_frag(FragBF& f, const __bf16* p) {
    f.h[0] = *(const v8bf*)(p);
    f.h[1] = *(const v8bf*)(p + 8);
}
__device__ inline void ld_fragA(FragBF& f, const __bf16* p) {  // A: one contiguous run of 16
    f.h[0] = *(const v8bf*)(p);
    f.h[1] = *(const v8bf*)(p + 8);
}

// ---------------------------------------------------------------------------
// Rolled, software-pipelined 16x16 GEMM pair: c0 += A x B(col0), c1 += A x B(col1).
// bA0/bA1 are loop-carried (loaded at bottom for next iteration); bB0/bB1 are
// iteration-local. 4 global_load_b128 stay in flight across every WMMA pair.
// ---------------------------------------------------------------------------
__device__ inline void gemm_pair(const __bf16* aRow, const __bf16* p0,
                                 const __bf16* p1, v8f& c0, v8f& c1) {
    FragBF bA0, bA1, bB0, bB1, a0, a1;
    ld_frag(bA0, p0);
    ld_frag(bA1, p1);
    int kb = 0;
#pragma unroll 1
    for (; kb < DD - 64; kb += 64) {
        ld_frag(bB0, p0 + kb + 32);          // this iteration, 2nd half
        ld_frag(bB1, p1 + kb + 32);
        ld_fragA(a0, aRow + kb);
        ld_fragA(a1, aRow + kb + 32);
        c0 = wmma_bf16(a0, bA0, c0);
        c1 = wmma_bf16(a0, bA1, c1);
        ld_frag(bA0, p0 + kb + 64);          // next iteration, 1st half
        ld_frag(bA1, p1 + kb + 64);
        c0 = wmma_bf16(a1, bB0, c0);
        c1 = wmma_bf16(a1, bB1, c1);
    }
    // tail: kb == DD-64, no next-chunk loads
    ld_frag(bB0, p0 + kb + 32);
    ld_frag(bB1, p1 + kb + 32);
    ld_fragA(a0, aRow + kb);
    ld_fragA(a1, aRow + kb + 32);
    c0 = wmma_bf16(a0, bA0, c0);
    c1 = wmma_bf16(a0, bA1, c1);
    c0 = wmma_bf16(a1, bB0, c0);
    c1 = wmma_bf16(a1, bB1, c1);
}

// ---------------------------------------------------------------------------
// Kernel 1: class weights cw[c] = (sum(counts)/counts[c]) / L1norm
// ---------------------------------------------------------------------------
__global__ void cw_kernel(const float* __restrict__ counts, float* __restrict__ cw) {
    if (threadIdx.x == 0) {
        float s = 0.f;
        for (int c = 0; c < NCLS; ++c) s += counts[c];
        float w[NCLS];
        float t = 0.f;
        for (int c = 0; c < NCLS; ++c) { w[c] = s / counts[c]; t += fabsf(w[c]); }
        for (int c = 0; c < NCLS; ++c) cw[c] = w[c] / t;
    }
}

// ---------------------------------------------------------------------------
// Kernel 2: L2-normalize each row, convert to bf16. One block per row.
// ---------------------------------------------------------------------------
__global__ __launch_bounds__(256) void norm_kernel(const float* __restrict__ f,
                                                   __bf16* __restrict__ fb) {
    __shared__ float red[256];
    const int row = blockIdx.x;
    const int tid = threadIdx.x;
    const float* src = f + (size_t)row * DD;

    float ss = 0.f;
    for (int d = tid; d < DD; d += 256) { float v = src[d]; ss += v * v; }
    red[tid] = ss;
    __syncthreads();
    for (int off = 128; off > 0; off >>= 1) {
        if (tid < off) red[tid] += red[tid + off];
        __syncthreads();
    }
    const float inv = rsqrtf(red[0]);
    __bf16* dst = fb + (size_t)row * DD;
    for (int d = tid; d < DD; d += 256) dst[d] = (__bf16)(src[d] * inv);
}

// ---------------------------------------------------------------------------
// Stage this block's 16xDD bf16 A-tile into padded LDS.
// Prefers gfx1250 GLOBAL_LOAD_ASYNC_TO_LDS_B128 (ASYNCcnt path, no VGPR data).
// ---------------------------------------------------------------------------
__device__ inline void load_a_tile(const __bf16* __restrict__ fb, int rbase,
                                   __bf16* lA, int tid) {
#ifdef USE_ASYNC_LDS
    for (int i = tid; i < 16 * (DD / 8); i += 256) {
        const int r = i >> 7;            // / 128
        const int c = (i & 127) << 3;    // * 8
        __builtin_amdgcn_global_load_async_to_lds_b128(
            (GLOBAL_AS v4i*)(fb + (size_t)(rbase + r) * DD + c),
            (LDS_AS v4i*)(lA + r * DDP + c), 0, 0);
    }
    __builtin_amdgcn_s_wait_asynccnt(0);
#else
    for (int i = tid; i < 16 * (DD / 8); i += 256) {
        const int r = i >> 7;
        const int c = (i & 127) << 3;
        *(uint4*)(lA + r * DDP + c) =
            *(const uint4*)(fb + (size_t)(rbase + r) * DD + c);
    }
#endif
}

// ---------------------------------------------------------------------------
// Kernel 3 (pass 1): per-row logsumexp of sim (diag masked to NEG_INF).
// One block = one 16-row stripe; 8 waves sweep 512 column tiles, 2 at a time.
// ---------------------------------------------------------------------------
__global__ __launch_bounds__(256) void pass1_kernel(const __bf16* __restrict__ fb,
                                                    float* __restrict__ logZ) {
    __shared__ __bf16 lA[16 * DDP];
    __shared__ float redM[8][16];
    __shared__ float redS[8][16];

    const int tid   = threadIdx.x;
    const int w     = tid >> 5;
    const int lane  = tid & 31;
    const int rbase = blockIdx.x * 16;

    load_a_tile(fb, rbase, lA, tid);
    __syncthreads();

    const int  mrow  = lane & 15;            // A: M index (ISA 16-bit A layout)
    const int  koffA = (lane & 16) ? 16 : 0;
    const int  koffB = (lane & 16) ? 16 : 0;
    const int  rhalf = (lane & 16) ? 8 : 0;  // C/D: M = r + rhalf
    const __bf16* aRow = lA + mrow * DDP + koffA;

    float m[8], s[8];
#pragma unroll
    for (int r = 0; r < 8; ++r) { m[r] = -__builtin_inff(); s[r] = 0.f; }

#pragma unroll 1
    for (int g = 0; g < 32; ++g) {           // 2 column tiles per iteration
        const int col0 = (w + 16 * g) * 16 + (lane & 15);
        const int col1 = col0 + 128;         // tile jt+8
        const __bf16* p0 = fb + (size_t)col0 * DD + koffB;
        const __bf16* p1 = fb + (size_t)col1 * DD + koffB;
        if (col0 + 256 < NN) {               // prefetch next pair's stripes
            __builtin_prefetch(fb + (size_t)(col0 + 256) * DD, 0, 1);
            __builtin_prefetch(fb + (size_t)(col1 + 256) * DD, 0, 1);
        }

        v8f c0 = {}, c1 = {};
        gemm_pair(aRow, p0, p1, c0, c1);

#pragma unroll
        for (int r = 0; r < 8; ++r) {
            const int row = rbase + r + rhalf;
            float v0 = c0[r] * INV_TEMP; if (row == col0) v0 = NEG_INF_V;
            float v1 = c1[r] * INV_TEMP; if (row == col1) v1 = NEG_INF_V;
            const float nm = fmaxf(m[r], fmaxf(v0, v1));
            s[r] = s[r] * __expf(m[r] - nm) + __expf(v0 - nm) + __expf(v1 - nm);
            m[r] = nm;
        }
    }

    // merge across the 16 lanes sharing each row (xor within half-wave)
#pragma unroll
    for (int off = 1; off < 16; off <<= 1) {
#pragma unroll
        for (int r = 0; r < 8; ++r) {
            const float om = __shfl_xor(m[r], off, 32);
            const float os = __shfl_xor(s[r], off, 32);
            const float nm = fmaxf(m[r], om);
            s[r] = s[r] * __expf(m[r] - nm) + os * __expf(om - nm);
            m[r] = nm;
        }
    }
    if ((lane & 15) == 0) {
#pragma unroll
        for (int r = 0; r < 8; ++r) { redM[w][rhalf + r] = m[r]; redS[w][rhalf + r] = s[r]; }
    }
    __syncthreads();

    if (tid < 16) {
        float M = -__builtin_inff(), S = 0.f;
        for (int ww = 0; ww < 8; ++ww) {
            const float om = redM[ww][tid], os = redS[ww][tid];
            const float nm = fmaxf(M, om);
            S = S * __expf(M - nm) + os * __expf(om - nm);
            M = nm;
        }
        logZ[rbase + tid] = M + __logf(S);
    }
}

// ---------------------------------------------------------------------------
// Kernel 4 (pass 2): focal terms over positive pairs, per-block partial sum.
// ---------------------------------------------------------------------------
__global__ __launch_bounds__(256) void pass2_kernel(const __bf16* __restrict__ fb,
                                                    const int* __restrict__ labels,
                                                    const float* __restrict__ logZ,
                                                    const float* __restrict__ cw,
                                                    float* __restrict__ blockSum) {
    __shared__ __bf16 lA[16 * DDP];
    __shared__ float lz[16];
    __shared__ int   labr[16];
    __shared__ float redP[8][16];
    __shared__ float redC[8][16];
    __shared__ float contribs[16];

    const int tid   = threadIdx.x;
    const int w     = tid >> 5;
    const int lane  = tid & 31;
    const int rbase = blockIdx.x * 16;

    load_a_tile(fb, rbase, lA, tid);
    if (tid < 16) { lz[tid] = logZ[rbase + tid]; labr[tid] = labels[rbase + tid]; }
    __syncthreads();

    const int  mrow  = lane & 15;
    const int  koffA = (lane & 16) ? 16 : 0;
    const int  koffB = (lane & 16) ? 16 : 0;
    const int  rhalf = (lane & 16) ? 8 : 0;
    const __bf16* aRow = lA + mrow * DDP + koffA;

    float lzr[8]; int lbr[8];
#pragma unroll
    for (int r = 0; r < 8; ++r) { lzr[r] = lz[r + rhalf]; lbr[r] = labr[r + rhalf]; }

    float ps[8], pc[8];
#pragma unroll
    for (int r = 0; r < 8; ++r) { ps[r] = 0.f; pc[r] = 0.f; }

#pragma unroll 1
    for (int g = 0; g < 32; ++g) {
        const int col0 = (w + 16 * g) * 16 + (lane & 15);
        const int col1 = col0 + 128;
        const int labc0 = labels[col0];
        const int labc1 = labels[col1];
        const __bf16* p0 = fb + (size_t)col0 * DD + koffB;
        const __bf16* p1 = fb + (size_t)col1 * DD + koffB;
        if (col0 + 256 < NN) {
            __builtin_prefetch(fb + (size_t)(col0 + 256) * DD, 0, 1);
            __builtin_prefetch(fb + (size_t)(col1 + 256) * DD, 0, 1);
        }

        v8f c0 = {}, c1 = {};
        gemm_pair(aRow, p0, p1, c0, c1);

#pragma unroll
        for (int r = 0; r < 8; ++r) {
            const int row = rbase + r + rhalf;
            if (row != col0 && lbr[r] == labc0) {
                const float lp = c0[r] * INV_TEMP - lzr[r];
                const float p  = __expf(lp);
                const float om = 1.0f - p;
                ps[r] += lp * om * om;
                pc[r] += 1.0f;
            }
            if (row != col1 && lbr[r] == labc1) {
                const float lp = c1[r] * INV_TEMP - lzr[r];
                const float p  = __expf(lp);
                const float om = 1.0f - p;
                ps[r] += lp * om * om;
                pc[r] += 1.0f;
            }
        }
    }

#pragma unroll
    for (int off = 1; off < 16; off <<= 1) {
#pragma unroll
        for (int r = 0; r < 8; ++r) {
            ps[r] += __shfl_xor(ps[r], off, 32);
            pc[r] += __shfl_xor(pc[r], off, 32);
        }
    }
    if ((lane & 15) == 0) {
#pragma unroll
        for (int r = 0; r < 8; ++r) { redP[w][rhalf + r] = ps[r]; redC[w][rhalf + r] = pc[r]; }
    }
    __syncthreads();

    if (tid < 16) {
        float P = 0.f, C = 0.f;
        for (int ww = 0; ww < 8; ++ww) { P += redP[ww][tid]; C += redC[ww][tid]; }
        const float per = (C > 0.f) ? (P / C) : 0.f;
        contribs[tid] = per * cw[labr[tid]];
    }
    __syncthreads();
    if (tid == 0) {
        float t = 0.f;
        for (int i = 0; i < 16; ++i) t += contribs[i];
        blockSum[blockIdx.x] = t;
    }
}

// ---------------------------------------------------------------------------
// Kernel 5: reduce 512 block partials -> loss = -sum / N
// ---------------------------------------------------------------------------
__global__ __launch_bounds__(256) void final_kernel(const float* __restrict__ blockSum,
                                                    float* __restrict__ out) {
    __shared__ float red[256];
    const int tid = threadIdx.x;
    red[tid] = blockSum[tid] + blockSum[tid + 256];
    __syncthreads();
    for (int off = 128; off > 0; off >>= 1) {
        if (tid < off) red[tid] += red[tid + off];
        __syncthreads();
    }
    if (tid == 0) out[0] = -red[0] / (float)NN;
}

// ---------------------------------------------------------------------------
extern "C" void kernel_launch(void* const* d_in, const int* in_sizes, int n_in,
                              void* d_out, int out_size, void* d_ws, size_t ws_size,
                              hipStream_t stream) {
    const float* features = (const float*)d_in[0];
    const int*   labels   = (const int*)d_in[1];
    const float* counts   = (const float*)d_in[2];
    float*       out      = (float*)d_out;

    // workspace layout
    char*   ws       = (char*)d_ws;
    __bf16* fb       = (__bf16*)ws;                                     // 16 MB
    float*  logZ     = (float*)(ws + (size_t)NN * DD * sizeof(__bf16)); // 32 KB
    float*  cw       = logZ + NN;                                       // 7 (+pad)
    float*  blockSum = cw + 16;                                         // 512

    cw_kernel   <<<1,     32,  0, stream>>>(counts, cw);
    norm_kernel <<<NN,    256, 0, stream>>>(features, fb);
    pass1_kernel<<<NN/16, 256, 0, stream>>>(fb, logZ);
    pass2_kernel<<<NN/16, 256, 0, stream>>>(fb, labels, logZ, cw, blockSum);
    final_kernel<<<1,     256, 0, stream>>>(blockSum, out);
}
